// NeuralUniLasso_26061861552586
// MI455X (gfx1250) — compile-verified, matrix operand order checked by reading, not run on hardware
//
#include <hip/hip_runtime.h>
#include <math.h>

// MI455X / gfx1250: wave32, WMMA 16x16x32 f16 -> f32 accum.
typedef __attribute__((ext_vector_type(16))) _Float16 v16h;
typedef __attribute__((ext_vector_type(8)))  float    v8f;

#define B_ROWS 32768
#define F_DIM  512
#define H1D    12
#define H2D    8
#define WAVES_PER_BLOCK 8
#define TILES_PER_WAVE  4
#define TILE_M 16

// Hardware tanh: single V_TANH_F32 trans op (confirmed in disasm).
static __device__ __forceinline__ float tanh_hw(float x) {
#if __has_builtin(__builtin_amdgcn_tanhf)
  return __builtin_amdgcn_tanhf(x);
#else
  float e = __builtin_amdgcn_exp2f(x * -2.885390081777927f);
  return (1.0f - e) * __builtin_amdgcn_rcpf(1.0f + e);
#endif
}

// Butterfly add over 8-lane groups using DPP8 (pure VALU, no LDS traffic).
constexpr unsigned dpp8_sel_xor(int m) {
  unsigned s = 0;
  for (int i = 0; i < 8; ++i) s |= (unsigned)((i ^ m) & 7) << (3 * i);
  return s;
}
template <unsigned SEL>
static __device__ __forceinline__ float dpp8_xor_add(float v) {
#if __has_builtin(__builtin_amdgcn_mov_dpp8)
  int p = __builtin_amdgcn_mov_dpp8(__float_as_int(v), SEL);
  return v + __int_as_float(p);
#else
  return v + __shfl_xor(v, (SEL & 7));
#endif
}
static __device__ __forceinline__ float group8_reduce(float v) {
  v = dpp8_xor_add<dpp8_sel_xor(1)>(v);
  v = dpp8_xor_add<dpp8_sel_xor(2)>(v);
  v = dpp8_xor_add<dpp8_sel_xor(4)>(v);
  return v;  // valid on lanes where (lane & 7) == 0
}

// ---------------------------------------------------------------------------
// Kernel 1: weights[f] = softplus(theta[f])
// ---------------------------------------------------------------------------
__global__ __launch_bounds__(256) void softplus_kernel(
    const float* __restrict__ theta, float* __restrict__ weights) {
  int f = blockIdx.x * blockDim.x + threadIdx.x;
  if (f < F_DIM) {
    float t = theta[f];
    weights[f] = (t > 20.f) ? t : log1pf(expf(t));
  }
}

// ---------------------------------------------------------------------------
// Kernel 2: Z[b,f]. One wave = TWO features (f0, f0+1) x 16-batch-row tile.
// Both 12x8 grouped GEMMs share one v_wmma_f32_16x16x32_f16 via a
// block-diagonal packing:
//   A: K 0..11  = h1(f0),  K 16..27 = h1(f1)          (zeros elsewhere)
//   B: (K0..11 x N0..7) = W2(f0), (K16..27 x N8..15) = W2(f1), else 0
//   => D cols 0..7 = h2pre(f0), cols 8..15 = h2pre(f1); all 32 lanes useful.
// Software-pipelined: tile t+1's A-fragment (16 v_tanh + FMAs, co-executable
// TRANS/VALU work) is built between WMMA t issue and its first consumer, so
// the f16-WMMA RAW hazard (5 co-exec slots) is covered by useful work.
//
// WMMA layouts (CDNA5 ISA 7.12.2, wave32):
//   A (16x32 f16): lane L -> row M=L&15; lanes 0-15: elems[0..7]=K0..7,
//     elems[8..15]=K16..23; lanes 16-31: elems[0..7]=K8..15, [8..15]=K24..31.
//   B (32x16 f16): lane L -> col N=L&15; same K striping as A.
//   C/D (16x16 f32): lane L -> col N=L&15, vgpr r -> row M=r+8*(L>=16).
// ---------------------------------------------------------------------------
__global__ __launch_bounds__(256) void z_kernel(
    const float* __restrict__ x,
    const float* __restrict__ W1, const float* __restrict__ b1,
    const float* __restrict__ W2, const float* __restrict__ b2,
    const float* __restrict__ W3, const float* __restrict__ b3,
    float* __restrict__ Z) {
  const int f0 = blockIdx.y * 2;

  // Zero-padded params for the feature pair.
  __shared__ float    sW1p[2][16], sb1p[2][16];  // K 12..15 = 0
  __shared__ _Float16 sW2h[32][16];              // block-diagonal, padded
  __shared__ float    sb2p[16], sW3p[16];        // [0..7]=f0, [8..15]=f1
  __shared__ float    sb3[2];

  const int tid = threadIdx.x;
  for (int idx = tid; idx < 512; idx += 256) {
    const int kk = idx >> 4, nn = idx & 15;
    float v = 0.f;
    if (nn < H2D) {                      // f0 block: K 0..11, N 0..7
      if (kk < H1D) v = W2[((size_t)f0 * H1D + kk) * H2D + nn];
    } else {                             // f1 block: K 16..27, N 8..15
      const int k2 = kk - 16;
      if (k2 >= 0 && k2 < H1D)
        v = W2[((size_t)(f0 + 1) * H1D + k2) * H2D + (nn - 8)];
    }
    sW2h[kk][nn] = (_Float16)v;
  }
  if (tid < 32) {
    const int feat = tid >> 4, k = tid & 15;
    const int ff = f0 + feat;
    const bool val = k < H1D;
    sW1p[feat][k] = val ? W1[ff * H1D + k] : 0.f;
    sb1p[feat][k] = val ? b1[ff * H1D + k] : 0.f;
  } else if (tid < 48) {
    const int n2 = tid - 32, feat = n2 >> 3, j = n2 & 7;
    const int ff = f0 + feat;
    sb2p[n2] = b2[ff * H2D + j];
    sW3p[n2] = W3[ff * H2D + j];
  } else if (tid < 50) {
    sb3[tid - 48] = b3[f0 + (tid - 48)];
  }
  __syncthreads();

  const int lane = tid & 31;
  const int wave = tid >> 5;
  const int n    = lane & 15;         // B/D column; also batch-row slot M=n
  const int kb   = (lane >> 4) << 3;  // K sub-base: 0 (lanes<16) or 8

  // B fragment (branch-free from padded LDS), built once per wave.
  v16h bfrag = {};
#pragma unroll
  for (int e = 0; e < 8; ++e) {
    bfrag[e]     = sW2h[kb + e][n];        // K = kb+e      (f0 block rows)
    bfrag[8 + e] = sW2h[16 + kb + e][n];   // K = 16+kb+e   (f1 block rows)
  }

  const float w3n = sW3p[n];
  const float b2n = sb2p[n];
  const float bb3 = sb3[(n >> 3) & 1];

  const int tile0 = blockIdx.x * (TILES_PER_WAVE * WAVES_PER_BLOCK) + wave;

  // Stage 1: issue all x loads up front (independent global_load_b64s).
  float2 xv[TILES_PER_WAVE];
#pragma unroll
  for (int t = 0; t < TILES_PER_WAVE; ++t) {
    const int tile_base = (tile0 + t * WAVES_PER_BLOCK) * TILE_M;
    xv[t] = *(const float2*)&x[(size_t)(tile_base + n) * F_DIM + f0];
  }

  // A-fragment builder: h1 = tanh(x*W1+b1) for both features (padded -> 0).
  auto build_afrag = [&](float2 xv2) {
    v16h a = {};
#pragma unroll
    for (int e = 0; e < 8; ++e) {
      a[e]     = (_Float16)tanh_hw(fmaf(xv2.x, sW1p[0][kb + e], sb1p[0][kb + e]));
      a[8 + e] = (_Float16)tanh_hw(fmaf(xv2.y, sW1p[1][kb + e], sb1p[1][kb + e]));
    }
    return a;
  };

  v16h afrag = build_afrag(xv[0]);

#pragma unroll
  for (int t = 0; t < TILES_PER_WAVE; ++t) {
    const int tile_base = (tile0 + t * WAVES_PER_BLOCK) * TILE_M;

    v8f c = {};
    c = __builtin_amdgcn_wmma_f32_16x16x32_f16(
        false, afrag, false, bfrag, (short)0, c, false, false);

    // Overlap: build next tile's A fragment while the WMMA executes.
    if (t + 1 < TILES_PER_WAVE) afrag = build_afrag(xv[t + 1]);

    // h2 = tanh(c + b2); partial = h2 * W3; reduce over each 8-lane N-group.
    float zr[8];
#pragma unroll
    for (int r = 0; r < 8; ++r)
      zr[r] = group8_reduce(tanh_hw(c[r] + b2n) * w3n);

    // Leaders: lane 0 (f0,M=0..7), 8 (f1,M=0..7), 16 (f0,M=8..15), 24 (f1,M=8..15)
    if ((lane & 7) == 0) {
      const int feat  = (lane >> 3) & 1;
      const int mbase = (lane >> 4) << 3;
#pragma unroll
      for (int r = 0; r < 8; ++r)
        Z[(size_t)(tile_base + mbase + r) * F_DIM + f0 + feat] = zr[r] + bb3;
    }
  }
}

// ---------------------------------------------------------------------------
// Kernel 3: out[b] = dot(Z[b,:], weights) + bias. One 256-thread block / row.
// ---------------------------------------------------------------------------
__global__ __launch_bounds__(256) void out_kernel(
    const float* __restrict__ Z, const float* __restrict__ w,
    const float* __restrict__ bias, float* __restrict__ out) {
  const int b = blockIdx.x;
  const int t = threadIdx.x;
  const float* zr = Z + (size_t)b * F_DIM;

  float s = fmaf(zr[t], w[t], zr[t + 256] * w[t + 256]);
  s = group8_reduce(s);
  s += __shfl_xor(s, 8);
  s += __shfl_xor(s, 16);

  __shared__ float ws[8];
  if ((t & 31) == 0) ws[t >> 5] = s;
  __syncthreads();
  if (t < 8) {
    float v = ws[t];
    v = group8_reduce(v);
    if (t == 0) out[b] = v + bias[0];
  }
}

// ---------------------------------------------------------------------------
// Launch: d_out = [ out (32768 f32) | weights (512 f32) | Z (32768*512 f32) ]
// ---------------------------------------------------------------------------
extern "C" void kernel_launch(void* const* d_in, const int* in_sizes, int n_in,
                              void* d_out, int out_size, void* d_ws, size_t ws_size,
                              hipStream_t stream) {
  (void)in_sizes; (void)n_in; (void)out_size; (void)d_ws; (void)ws_size;
  const float* x     = (const float*)d_in[0];
  const float* W1    = (const float*)d_in[1];
  const float* b1    = (const float*)d_in[2];
  const float* W2    = (const float*)d_in[3];
  const float* b2    = (const float*)d_in[4];
  const float* W3    = (const float*)d_in[5];
  const float* b3    = (const float*)d_in[6];
  const float* theta = (const float*)d_in[7];
  const float* bias  = (const float*)d_in[8];

  float* out     = (float*)d_out;
  float* weights = out + B_ROWS;
  float* Z       = weights + F_DIM;

  softplus_kernel<<<(F_DIM + 255) / 256, 256, 0, stream>>>(theta, weights);

  // batch tiles per feature-pair = B/16 = 2048 = grid.x(64) * waves(8) * tiles(4)
  dim3 zgrid(B_ROWS / (TILE_M * WAVES_PER_BLOCK * TILES_PER_WAVE), F_DIM / 2);
  z_kernel<<<zgrid, 256, 0, stream>>>(x, W1, b1, W2, b2, W3, b3, Z);

  out_kernel<<<B_ROWS, 256, 0, stream>>>(Z, weights, bias, out);
}
